// MLP_19645180412051
// MI455X (gfx1250) — compile-verified, matrix operand order checked by reading, not run on hardware
//
#include <hip/hip_runtime.h>
#include <hip/hip_bf16.h>

#define NN      100000     // nodes
#define NE      600000     // edges
#define CIN     256
#define CH      128
#define NTILES  (NN / 16)  // 6250 row tiles of 16
#define NEG_SLOPE 0.01f
#define LN_EPS    1e-5f

typedef __bf16 v16bf __attribute__((ext_vector_type(16)));
typedef float  v8f   __attribute__((ext_vector_type(8)));

__device__ __forceinline__ __bf16 f2bf(float f) {
    unsigned u = __float_as_uint(f);
    u += 0x7FFFu + ((u >> 16) & 1u);           // round-to-nearest-even
    unsigned short hs = (unsigned short)(u >> 16);
    return __builtin_bit_cast(__bf16, hs);
}

// ---------------------------------------------------------------------------
// Pack a [Ksteps*32, 128] f32 weight matrix into WMMA B-fragment bf16 layout.
// Fragment (kk, nt): lane L<16 -> col n = nt*16+L, elems j: K = kk*32 + {j | 16+j-8}
//                    lane L>=16 -> same cols, K offset +8 (second half of K pairs)
// ---------------------------------------------------------------------------
__global__ void pack_w_kernel(const float* __restrict__ W, __bf16* __restrict__ out,
                              int ksteps) {
    int idx = blockIdx.x * blockDim.x + threadIdx.x;
    int total = ksteps * 8 * 32 * 16;
    if (idx >= total) return;
    int j    = idx & 15;
    int lane = (idx >> 4) & 31;
    int frag = idx >> 9;            // kk*8 + nt
    int nt   = frag & 7;
    int kk   = frag >> 3;
    int half = lane >> 4;
    int n    = nt * 16 + (lane & 15);
    int koff = (j < 8) ? (half * 8 + j) : (16 + half * 8 + (j - 8));
    int k    = kk * 32 + koff;
    out[idx] = f2bf(W[k * CH + n]);
}

// ---------------------------------------------------------------------------
// agg init to +inf (bit pattern 0x7F800000)
// ---------------------------------------------------------------------------
__global__ void init_agg_kernel(unsigned int* __restrict__ agg) {
    size_t i = (size_t)blockIdx.x * blockDim.x + threadIdx.x;
    if (i < (size_t)NN * CH) agg[i] = 0x7F800000u;
}

// ---------------------------------------------------------------------------
// GEMM1 + bias + LeakyReLU + LayerNorm, fully fused per 16x128 tile per wave.
// Output h is stored in bf16: rounding is monotone, so the downstream
// scatter-min over bf16 values equals bf16(min over f32 values), which is
// exactly what GEMM2 would have computed anyway (it rounds A to bf16).
// ---------------------------------------------------------------------------
__global__ __launch_bounds__(256) void gemm1_ln_kernel(
    const float* __restrict__ x, const __bf16* __restrict__ w1p,
    const float* __restrict__ b1, const float* __restrict__ gamma,
    const float* __restrict__ beta, __bf16* __restrict__ h) {
    const int lane = threadIdx.x & 31;
    const int wid  = blockIdx.x * (blockDim.x >> 5) + (threadIdx.x >> 5);
    if (wid >= NTILES) return;                 // wave-uniform: EXEC stays all-1s
    const int row0 = wid * 16;
    const int half = lane >> 4;
    const int l16  = lane & 15;

    v8f acc[8];
    #pragma unroll
    for (int t = 0; t < 8; ++t)
        #pragma unroll
        for (int r = 0; r < 8; ++r) acc[t][r] = 0.0f;

    const float* xrow = x + (size_t)(row0 + l16) * CIN;

    for (int kk = 0; kk < 8; ++kk) {           // K = 256 -> 8 steps of 32
        const float4* px = (const float4*)(xrow + kk * 32 + half * 8);
        float4 f0 = px[0], f1 = px[1], f2 = px[4], f3 = px[5];
        v16bf a;
        a[0]=f2bf(f0.x); a[1]=f2bf(f0.y); a[2]=f2bf(f0.z); a[3]=f2bf(f0.w);
        a[4]=f2bf(f1.x); a[5]=f2bf(f1.y); a[6]=f2bf(f1.z); a[7]=f2bf(f1.w);
        a[8]=f2bf(f2.x); a[9]=f2bf(f2.y); a[10]=f2bf(f2.z); a[11]=f2bf(f2.w);
        a[12]=f2bf(f3.x); a[13]=f2bf(f3.y); a[14]=f2bf(f3.z); a[15]=f2bf(f3.w);

        const v16bf* bp = (const v16bf*)w1p + (size_t)(kk * 8) * 32 + lane;
        #pragma unroll
        for (int nt = 0; nt < 8; ++nt) {
            v16bf b = bp[nt * 32];
            acc[nt] = __builtin_amdgcn_wmma_f32_16x16x32_bf16(
                false, a, false, b, (short)0, acc[nt], false, false);
        }
    }

    // bias + LeakyReLU in registers
    #pragma unroll
    for (int nt = 0; nt < 8; ++nt) {
        float bb = b1[nt * 16 + l16];
        #pragma unroll
        for (int r = 0; r < 8; ++r) {
            float v = acc[nt][r] + bb;
            acc[nt][r] = (v > 0.0f) ? v : NEG_SLOPE * v;
        }
    }

    // LayerNorm: VGPR r holds row (row0 + half*8 + r), cols spread over the
    // 16 lanes of this half and the 8 column tiles -> in-lane + shfl reduce.
    float mu[8], rs[8];
    #pragma unroll
    for (int r = 0; r < 8; ++r) {
        float s = 0.0f, q = 0.0f;
        #pragma unroll
        for (int nt = 0; nt < 8; ++nt) { float v = acc[nt][r]; s += v; q += v * v; }
        #pragma unroll
        for (int m = 1; m < 16; m <<= 1) {
            s += __shfl_xor(s, m, 32);
            q += __shfl_xor(q, m, 32);
        }
        float mean = s * (1.0f / 128.0f);
        float var  = q * (1.0f / 128.0f) - mean * mean;
        mu[r] = mean;
        rs[r] = rsqrtf(var + LN_EPS);
    }

    #pragma unroll
    for (int nt = 0; nt < 8; ++nt) {
        float g  = gamma[nt * 16 + l16];
        float be = beta[nt * 16 + l16];
        #pragma unroll
        for (int r = 0; r < 8; ++r) {
            float v = (acc[nt][r] - mu[r]) * rs[r] * g + be;
            h[(size_t)(row0 + half * 8 + r) * CH + nt * 16 + l16] = f2bf(v);
        }
    }
}

// ---------------------------------------------------------------------------
// Edge scatter-min: one wave per edge, 4 channels per lane. h row is bf16
// (256 B / row -> one coalesced uint2 per lane), unpacked exactly to f32
// (u << 16), then sign-aware integer atomic min (init is +inf; the int view
// of IEEE floats is order-monotone for >=0, reversed for <0).
// ---------------------------------------------------------------------------
__device__ __forceinline__ void atomic_fmin(float* addr, float val) {
    if (val >= 0.0f) atomicMin((int*)addr, __float_as_int(val));
    else             atomicMax((unsigned int*)addr, __float_as_uint(val));
}

__global__ __launch_bounds__(256) void scatter_min_kernel(
    const int* __restrict__ ei, const __bf16* __restrict__ h,
    float* __restrict__ agg) {
    int wid  = blockIdx.x * (blockDim.x >> 5) + (threadIdx.x >> 5);
    int lane = threadIdx.x & 31;
    if (wid >= NE) return;
    int s = ei[wid];
    int d = ei[NE + wid];
    uint2 p = ((const uint2*)(h + (size_t)s * CH))[lane];
    float v0 = __uint_as_float(p.x << 16);            // col lane*4 + 0
    float v1 = __uint_as_float(p.x & 0xFFFF0000u);    // col lane*4 + 1
    float v2 = __uint_as_float(p.y << 16);            // col lane*4 + 2
    float v3 = __uint_as_float(p.y & 0xFFFF0000u);    // col lane*4 + 3
    float* ap = agg + (size_t)d * CH + lane * 4;
    atomic_fmin(ap + 0, v0);
    atomic_fmin(ap + 1, v1);
    atomic_fmin(ap + 2, v2);
    atomic_fmin(ap + 3, v3);
}

// ---------------------------------------------------------------------------
// GEMM2: out = mask(agg) @ W2 + b2   (inf -> 0 masking fused in A-load;
// a node row is either fully written by edges or fully +inf)
// ---------------------------------------------------------------------------
__device__ __forceinline__ float mask_inf(float v) {
    return (__float_as_uint(v) == 0x7F800000u) ? 0.0f : v;
}

__global__ __launch_bounds__(256) void gemm2_kernel(
    const float* __restrict__ agg, const __bf16* __restrict__ w2p,
    const float* __restrict__ b2, float* __restrict__ out) {
    const int lane = threadIdx.x & 31;
    const int wid  = blockIdx.x * (blockDim.x >> 5) + (threadIdx.x >> 5);
    if (wid >= NTILES) return;
    const int row0 = wid * 16;
    const int half = lane >> 4;
    const int l16  = lane & 15;

    v8f acc[8];
    #pragma unroll
    for (int t = 0; t < 8; ++t)
        #pragma unroll
        for (int r = 0; r < 8; ++r) acc[t][r] = 0.0f;

    const float* arow = agg + (size_t)(row0 + l16) * CH;

    #pragma unroll
    for (int kk = 0; kk < 4; ++kk) {           // K = 128 -> 4 steps of 32
        const float4* px = (const float4*)(arow + kk * 32 + half * 8);
        float4 f0 = px[0], f1 = px[1], f2 = px[4], f3 = px[5];
        v16bf a;
        a[0]=f2bf(mask_inf(f0.x)); a[1]=f2bf(mask_inf(f0.y));
        a[2]=f2bf(mask_inf(f0.z)); a[3]=f2bf(mask_inf(f0.w));
        a[4]=f2bf(mask_inf(f1.x)); a[5]=f2bf(mask_inf(f1.y));
        a[6]=f2bf(mask_inf(f1.z)); a[7]=f2bf(mask_inf(f1.w));
        a[8]=f2bf(mask_inf(f2.x)); a[9]=f2bf(mask_inf(f2.y));
        a[10]=f2bf(mask_inf(f2.z)); a[11]=f2bf(mask_inf(f2.w));
        a[12]=f2bf(mask_inf(f3.x)); a[13]=f2bf(mask_inf(f3.y));
        a[14]=f2bf(mask_inf(f3.z)); a[15]=f2bf(mask_inf(f3.w));

        const v16bf* bp = (const v16bf*)w2p + (size_t)(kk * 8) * 32 + lane;
        #pragma unroll
        for (int nt = 0; nt < 8; ++nt) {
            v16bf b = bp[nt * 32];
            acc[nt] = __builtin_amdgcn_wmma_f32_16x16x32_bf16(
                false, a, false, b, (short)0, acc[nt], false, false);
        }
    }

    #pragma unroll
    for (int nt = 0; nt < 8; ++nt) {
        float bb = b2[nt * 16 + l16];
        #pragma unroll
        for (int r = 0; r < 8; ++r) {
            out[(size_t)(row0 + half * 8 + r) * CH + nt * 16 + l16] =
                acc[nt][r] + bb;
        }
    }
}

// ---------------------------------------------------------------------------
// Host-side launcher.  Workspace layout (requires ~77 MB):
//   [0, 25600000)            h   : bf16 [NN,128]
//   [25600000, 76800000)     agg : f32  [NN,128]
//   [76800000, +64KB)        w1p : bf16 WMMA-fragment-packed W1
//   [+64KB, +96KB)           w2p : bf16 WMMA-fragment-packed W2
// ---------------------------------------------------------------------------
extern "C" void kernel_launch(void* const* d_in, const int* in_sizes, int n_in,
                              void* d_out, int out_size, void* d_ws, size_t ws_size,
                              hipStream_t stream) {
    const float* x     = (const float*)d_in[0];
    // d_in[1], d_in[2]: x_struct, x_e (unused by forward)
    const int*   ei    = (const int*)d_in[3];    // [2, NE]: row0=src, row1=dst
    const float* W1    = (const float*)d_in[4];
    const float* b1    = (const float*)d_in[5];
    const float* gamma = (const float*)d_in[6];
    const float* beta  = (const float*)d_in[7];
    const float* W2    = (const float*)d_in[8];
    const float* b2    = (const float*)d_in[9];
    float*       out   = (float*)d_out;

    char*   ws  = (char*)d_ws;
    __bf16* h   = (__bf16*)(ws);
    float*  agg = (float*)(ws + 25600000);
    __bf16* w1p = (__bf16*)(ws + 76800000);
    __bf16* w2p = (__bf16*)(ws + 76800000 + 65536);

    pack_w_kernel<<<(8 * 8 * 32 * 16 + 255) / 256, 256, 0, stream>>>(W1, w1p, 8);
    pack_w_kernel<<<(4 * 8 * 32 * 16 + 255) / 256, 256, 0, stream>>>(W2, w2p, 4);
    init_agg_kernel<<<(NN * CH) / 256, 256, 0, stream>>>((unsigned int*)agg);

    gemm1_ln_kernel<<<(NTILES + 7) / 8, 256, 0, stream>>>(x, w1p, b1, gamma, beta, h);
    scatter_min_kernel<<<NE / 8, 256, 0, stream>>>(ei, h, agg);
    gemm2_kernel<<<(NTILES + 7) / 8, 256, 0, stream>>>(agg, w2p, b2, out);
}
